// TwiceBlock_35287451304251
// MI455X (gfx1250) — matrix operand order repaired and checked by reading, not compiled, not run on hardware
//
#include <hip/hip_runtime.h>

// ---------------------------------------------------------------------------
// CDNA5 (gfx1250) wave32 WMMA TwiceBlock forward.
// All contractions: v_wmma_f32_16x16x32_bf16, operands pre-packed so every
// lane loads its 16-element fragment as one aligned 32B chunk (2x b128).
// GEMM hot loop is software-pipelined (double-buffered fragments, no branches).
// ---------------------------------------------------------------------------

typedef __attribute__((ext_vector_type(16))) __bf16 bf16x16;
typedef __attribute__((ext_vector_type(8)))  __bf16 bf16x8;
typedef __attribute__((ext_vector_type(8)))  float  f32x8;

#define BB   4
#define CC   256
#define HH   56
#define WW   56
#define NN   (HH * WW)          // 3136
#define HK   28
#define WK   28
#define NKV  (HK * WK)          // 784
#define NKP  800                // NKV padded to a multiple of 32
#define GG   4
#define CGC  64
#define HEADS 8
#define DH   32

__device__ __forceinline__ int imin(int a, int b) { return a < b ? a : b; }
__device__ __forceinline__ int imax(int a, int b) { return a > b ? a : b; }

__device__ __forceinline__ float gelu_f(float x) {
    float x3 = x * x * x;
    return 0.5f * x * (1.f + tanhf(0.7978845608028654f * (x + 0.044715f * x3)));
}

// K-mapping within a 32-block for A-operand packing (identity):
// packed position p holds original k = permk(p) = p (contiguous).
__device__ __forceinline__ int permk(int k) {
    return k;
}
// run base mapping for vectorized 8-wide stores
__device__ __forceinline__ int permrun(int x) {
    return x;
}

__device__ __forceinline__ f32x8 wmma_bf16(bf16x16 a, bf16x16 b, f32x8 c) {
    return __builtin_amdgcn_wmma_f32_16x16x32_bf16(false, a, false, b, (short)0, c,
                                                   false, false);
}

// ---------------------------------------------------------------------------
// Pack f32 weight [M,K] -> bf16 A-pack layout (per-32 K-permuted), row-major.
// ---------------------------------------------------------------------------
__global__ void k_packA(const float* __restrict__ W, __bf16* __restrict__ Wp,
                        int M, int K) {
    int t = blockIdx.x * blockDim.x + threadIdx.x;
    int kb32 = K >> 5;
    if (t >= M * kb32) return;
    int kb = t % kb32, m = t / kb32;
    const float* src = W + (size_t)m * K + kb * 32;
    __bf16* dst = Wp + (size_t)m * K + kb * 32;
#pragma unroll
    for (int p = 0; p < 32; ++p) dst[p] = (__bf16)src[permk(p)];
}

__global__ void k_zero(__bf16* __restrict__ p, int n) {
    int t = blockIdx.x * blockDim.x + threadIdx.x;
    if (t < n) p[t] = (__bf16)0.f;
}

// ---------------------------------------------------------------------------
// Channel LayerNorm over axis C of [Bn, C, N]; optional bf16 B-pack mirror.
// ---------------------------------------------------------------------------
__global__ void k_ln(const float* __restrict__ x, const float* __restrict__ g,
                     const float* __restrict__ bta, float* __restrict__ out,
                     __bf16* __restrict__ obf, int Bn, int C, int N) {
    int t = blockIdx.x * blockDim.x + threadIdx.x;
    if (t >= Bn * N) return;
    int b = t / N, n = t % N;
    const float* xp = x + (size_t)b * C * N + n;
    float s = 0.f, ss = 0.f;
    for (int c = 0; c < C; ++c) {
        float v = xp[(size_t)c * N];
        s += v; ss += v * v;
    }
    float mu  = s / C;
    float var = ss / C - mu * mu;
    float rs  = rsqrtf(var + 1e-5f);
    float* op = out + (size_t)b * C * N + n;
    __bf16* ob = obf ? obf + (size_t)b * C * N : nullptr;
    for (int c = 0; c < C; ++c) {
        float v = (xp[(size_t)c * N] - mu) * rs * g[c] + bta[c];
        op[(size_t)c * N] = v;
        if (ob) ob[((size_t)(c >> 5) * N + n) * 32 + (c & 31)] = (__bf16)v;
    }
}

// ---------------------------------------------------------------------------
// WMMA GEMM: out[z, c0+m, n] = act(Wp[m,:] . X[z,:,n] + bias[m]) (+res)
//   Wp : bf16 A-pack [M,K];  X : bf16 B-pack [z][K/32][N][32]
//   Y  : optional f32 [z][Mtot][N]
//   Ybf: optional bf16, bfmode 0 = B-pack, 1 = A-pack channel-last,
//                       2 = row-major with K(=n)-perm cols (ldn-padded)
//   grid=(ceil(N/64), ceil(M/64), batch), block=128; wave tile 16x64.
//   Hot path (full 64-col tile): double-buffered, branch-free K loop.
// ---------------------------------------------------------------------------
__global__ void k_gemm(const __bf16* __restrict__ Wp, const __bf16* __restrict__ X,
                       const float* __restrict__ bias, const float* __restrict__ res,
                       float* __restrict__ Y, __bf16* __restrict__ Ybf,
                       int M, int K, int N, int Mtot, int c0, int act,
                       int bfmode, int ldn) {
    const int lane = threadIdx.x & 31;
    const int wave = threadIdx.x >> 5;
    const int half = lane >> 4;
    const int lm   = lane & 15;
    const int bm   = blockIdx.y * 64 + wave * 16;   // wave-uniform
    if (bm >= M) return;
    const int bn = blockIdx.x * 64;
    const int z  = blockIdx.z;
    const __bf16* Xb = X + (size_t)z * K * N;

    f32x8 zero = {};
    f32x8 acc[4] = {zero, zero, zero, zero};

    const size_t xstep = (size_t)N * 32;
    const __bf16* wr = Wp + (size_t)(bm + lm) * K + half * 16;
    const __bf16* xp = Xb + (size_t)(bn + lm) * 32 + half * 16;
    const int kiters = K >> 5;
    const bool full = (bn + 64 <= N);
    const int jcnt = full ? 4 : ((N - bn) >> 4);    // wave-uniform

    if (full) {
        // -------- software-pipelined, branch-free hot loop --------
        bf16x16 a  = *(const bf16x16*)(wr);
        bf16x16 b0 = *(const bf16x16*)(xp);
        bf16x16 b1 = *(const bf16x16*)(xp + 512);
        bf16x16 b2 = *(const bf16x16*)(xp + 1024);
        bf16x16 b3 = *(const bf16x16*)(xp + 1536);
        for (int it = 1; it < kiters; ++it) {
            wr += 32; xp += xstep;
            bf16x16 an  = *(const bf16x16*)(wr);
            bf16x16 n0  = *(const bf16x16*)(xp);
            bf16x16 n1  = *(const bf16x16*)(xp + 512);
            bf16x16 n2  = *(const bf16x16*)(xp + 1024);
            bf16x16 n3  = *(const bf16x16*)(xp + 1536);
            acc[0] = wmma_bf16(a, b0, acc[0]);
            acc[1] = wmma_bf16(a, b1, acc[1]);
            acc[2] = wmma_bf16(a, b2, acc[2]);
            acc[3] = wmma_bf16(a, b3, acc[3]);
            a = an; b0 = n0; b1 = n1; b2 = n2; b3 = n3;
        }
        acc[0] = wmma_bf16(a, b0, acc[0]);
        acc[1] = wmma_bf16(a, b1, acc[1]);
        acc[2] = wmma_bf16(a, b2, acc[2]);
        acc[3] = wmma_bf16(a, b3, acc[3]);
    } else {
        // -------- cold remainder path (N = 784 tail tile) --------
        for (int it = 0; it < kiters; ++it) {
            bf16x16 a = *(const bf16x16*)(wr);
            for (int j = 0; j < jcnt; ++j) {
                bf16x16 bv = *(const bf16x16*)(xp + (size_t)j * 512);
                acc[j] = wmma_bf16(a, bv, acc[j]);
            }
            wr += 32; xp += xstep;
        }
    }

    const size_t zb = (size_t)z * Mtot * ldn;
    const int mg0 = c0 + bm;                 // multiple of 16 (c0 mult of 64)
    const int sub = (mg0 & 31) + 8 * half;   // run base within 32-block

#pragma unroll
    for (int j = 0; j < 4; ++j) {
        if (j < jcnt) {
            int n = bn + j * 16 + lm;
            float vv[8];
#pragma unroll
            for (int r = 0; r < 8; ++r) {
                int m = bm + r + 8 * half;
                float v = acc[j][r];
                if (bias) v += bias[m];
                if (act == 1) v = gelu_f(v);
                size_t oi = ((size_t)z * Mtot + c0 + m) * (size_t)N + n;
                if (res) v += res[oi];
                if (Y)   Y[oi] = v;
                vv[r] = v;
            }
            if (Ybf) {
                if (bfmode == 2) {
                    int npart = (n & ~31) + permk(n & 31);
#pragma unroll
                    for (int r = 0; r < 8; ++r)
                        Ybf[zb + (size_t)(mg0 + r + 8 * half) * ldn + npart] =
                            (__bf16)vv[r];
                } else {
                    bf16x8 pk;
#pragma unroll
                    for (int r = 0; r < 8; ++r) pk[r] = (__bf16)vv[r];
                    size_t idx0;
                    if (bfmode == 0)
                        idx0 = zb + ((size_t)(mg0 >> 5) * ldn + n) * 32 + sub;
                    else
                        idx0 = zb + (size_t)n * Mtot + (mg0 & ~31) + permrun(sub);
                    *(bf16x8*)(Ybf + idx0) = pk;
                }
            }
        }
    }
}

// ---------------------------------------------------------------------------
// Depthwise conv over [Bc, Ctot, H, W] channel slice [c0, c0+Cn).
// ---------------------------------------------------------------------------
__global__ void k_dw(const float* __restrict__ in, const float* __restrict__ wgt,
                     const float* __restrict__ bias, const float* __restrict__ res,
                     float* __restrict__ out, int Bc, int Ctot, int c0, int Cn,
                     int H, int W, int Ho, int Wo, int ksz, int stride, int pad,
                     int act) {
    int t = blockIdx.x * blockDim.x + threadIdx.x;
    int tot = Bc * Cn * Ho * Wo;
    if (t >= tot) return;
    int wo_ = t % Wo;
    int ho  = (t / Wo) % Ho;
    int c   = (t / (Wo * Ho)) % Cn;
    int bc  = t / (Wo * Ho * Cn);
    float sum = bias ? bias[c] : 0.f;
    const float* ip = in + ((size_t)bc * Ctot + c0 + c) * (size_t)(H * W);
    const float* wp = wgt + (size_t)c * ksz * ksz;
    for (int ky = 0; ky < ksz; ++ky) {
        int hi = ho * stride - pad + ky;
        if (hi < 0 || hi >= H) continue;
        for (int kx = 0; kx < ksz; ++kx) {
            int wi = wo_ * stride - pad + kx;
            if (wi < 0 || wi >= W) continue;
            sum += ip[hi * W + wi] * wp[ky * ksz + kx];
        }
    }
    if (act == 1) sum = gelu_f(sum);
    size_t oi = ((size_t)bc * Ctot + c0 + c) * (size_t)(Ho * Wo) + ho * Wo + wo_;
    if (res) sum += res[oi];
    out[oi] = sum;
}

__global__ void k_copyc(const float* __restrict__ src, float* __restrict__ dst,
                        int Bc, int Ctot, int c0, int Cn, int N) {
    int t = blockIdx.x * blockDim.x + threadIdx.x;
    int tot = Bc * Cn * N;
    if (t >= tot) return;
    int n = t % N, c = (t / N) % Cn, bc = t / (N * Cn);
    size_t idx = ((size_t)bc * Ctot + c0 + c) * (size_t)N + n;
    dst[idx] = src[idx];
}

__global__ void k_mean(const float* __restrict__ y, float* __restrict__ pooled,
                       int rows, int N) {
    int t = blockIdx.x * blockDim.x + threadIdx.x;
    if (t >= rows) return;
    const float* p = y + (size_t)t * N;
    float s = 0.f;
    for (int i = 0; i < N; ++i) s += p[i];
    pooled[t] = s / N;
}

__global__ void k_eca(const float* __restrict__ pooled, const float* __restrict__ w,
                      float* __restrict__ att) {
    int t = blockIdx.x * blockDim.x + threadIdx.x;
    if (t >= BB * CC) return;
    int b = t / CC, c = t % CC;
    float s = 0.f;
    for (int j = 0; j < 9; ++j) {
        int cc = c + j - 4;
        if (cc >= 0 && cc < CC) s += pooled[b * CC + cc] * w[j];
    }
    att[t] = 1.f / (1.f + __expf(-s));
}

// y *= att (att already holds the sigmoid); bf16 mirror in B-pack layout
__global__ void k_scale(float* __restrict__ y, const float* __restrict__ att,
                        __bf16* __restrict__ ybf, int N, int tot) {
    int t = blockIdx.x * blockDim.x + threadIdx.x;
    if (t >= tot) return;
    int r0 = t / N, n = t % N;
    int b = r0 >> 8, c = r0 & 255;
    float v = y[t] * att[r0];
    y[t] = v;
    ybf[(size_t)b * CC * N + ((size_t)(c >> 5) * N + n) * 32 + (c & 31)] = (__bf16)v;
}

__global__ void k_pwoff(const float* __restrict__ off1, const float* __restrict__ w,
                        const float* __restrict__ b, float* __restrict__ off2) {
    int t = blockIdx.x * blockDim.x + threadIdx.x;
    int tot = BB * GG * 2 * NKV;
    if (t >= tot) return;
    int p  = t % NKV;
    int o  = (t / NKV) % 2;
    int bg = t / (NKV * 2);
    float s = b[o];
    const float* ip = off1 + (size_t)bg * CGC * NKV + p;
    for (int c = 0; c < CGC; ++c) s += w[o * CGC + c] * ip[(size_t)c * NKV];
    off2[((size_t)bg * 2 + o) * NKV + p] = s;
}

// tanh offsets + ref grid + clamp + bilinear gather -> samp (bf16 B-pack)
__global__ void k_sample(const float* __restrict__ y, const float* __restrict__ off2,
                         __bf16* __restrict__ samp) {
    int t = blockIdx.x * blockDim.x + threadIdx.x;
    int tot = BB * GG * NKV;
    if (t >= tot) return;
    int p  = t % NKV;
    int bg = t / NKV;
    int b = bg >> 2, g = bg & 3;
    int i = p / WK, j = p % WK;
    float rx = -1.f + 2.f * j / (float)(WK - 1);
    float ry = -1.f + 2.f * i / (float)(HK - 1);
    float px_ = rx + tanhf(off2[((size_t)bg * 2 + 0) * NKV + p]);
    float py_ = ry + tanhf(off2[((size_t)bg * 2 + 1) * NKV + p]);
    px_ = fminf(fmaxf(px_, -1.f), 1.f);
    py_ = fminf(fmaxf(py_, -1.f), 1.f);
    float px = (px_ + 1.f) * 0.5f * (WW - 1);
    float py = (py_ + 1.f) * 0.5f * (HH - 1);
    float x0 = floorf(px), y0 = floorf(py);
    float wx = px - x0, wy = py - y0;
    int x0i = imin(imax((int)x0, 0), WW - 1);
    int x1i = imin(x0i + 1, WW - 1);
    int y0i = imin(imax((int)y0, 0), HH - 1);
    int y1i = imin(y0i + 1, HH - 1);
    float w00 = (1.f - wy) * (1.f - wx), w01 = (1.f - wy) * wx;
    float w10 = wy * (1.f - wx), w11 = wy * wx;
    const float* fb = y + (size_t)bg * CGC * NN;
    for (int c = 0; c < CGC; ++c) {
        const float* f = fb + (size_t)c * NN;
        float v = f[y0i * WW + x0i] * w00 + f[y0i * WW + x1i] * w01 +
                  f[y1i * WW + x0i] * w10 + f[y1i * WW + x1i] * w11;
        int ch = g * CGC + c;
        samp[(size_t)b * CC * NKV + ((size_t)(ch >> 5) * NKV + p) * 32 + (ch & 31)] =
            (__bf16)v;
    }
}

// ---------------------------------------------------------------------------
// Flash attention. One wave per (b, head, 16-query tile).
//   qa : bf16 A-pack channel-last [B][N][CC]
//   kp : bf16 B-pack [B][8][NKV][32]
//   va : bf16 row-major, key-permuted cols, padded [B][CC][NKP] (zero-filled)
//   obf: bf16 B-pack [B][8][N][32]
// ---------------------------------------------------------------------------
__global__ void k_attn(const __bf16* __restrict__ qa, const __bf16* __restrict__ kp,
                       const __bf16* __restrict__ va, __bf16* __restrict__ obf,
                       int N, int Nk) {
    const int lane = threadIdx.x & 31;
    const int wave = threadIdx.x >> 5;
    const int half = lane >> 4;
    const int nl   = lane & 15;
    const int qt   = blockIdx.x * 4 + wave;
    const int head = blockIdx.y, b = blockIdx.z;
    const int q0 = qt * 16;
    const __bf16* qb = qa + (size_t)b * N * CC;
    const __bf16* kb = kp + (size_t)b * CC * Nk + (size_t)head * Nk * 32;
    const __bf16* vb = va + (size_t)b * CC * NKP + (size_t)(head * DH) * NKP;

    __shared__ float sS[4 * 16 * 33];
    float* S = sS + wave * 528;

    const float scale = 0.17677669529663687f;   // 1/sqrt(32)
    float mrun = -1e30f, lrun = 0.f;
    f32x8 o0 = {}, o1 = {};

    // Q^T fragment: one aligned 32B load (A-pack, channel-last)
    bf16x16 aq = *(const bf16x16*)(qb + (size_t)(q0 + nl) * CC + head * DH + half * 16);

    for (int kb0 = 0; kb0 < Nk; kb0 += 32) {
        const int kvalid = imin(32, Nk - kb0);
#pragma unroll
        for (int j = 0; j < 2; ++j) {
            if (kb0 + j * 16 < Nk) {                      // wave-uniform
                int key = kb0 + j * 16 + nl;
                bf16x16 bk = *(const bf16x16*)(kb + (size_t)key * 32 + half * 16);
                f32x8 s = {};
                s = wmma_bf16(aq, bk, s);
#pragma unroll
                for (int r = 0; r < 8; ++r)
                    S[(r + 8 * half) * 33 + j * 16 + nl] = s[r];
            }
        }
        asm volatile("s_wait_dscnt 0" ::: "memory");

        // per-lane online softmax for query nl
        float rmax = -1e30f;
        for (int t = 0; t < 32; ++t)
            if (t < kvalid) rmax = fmaxf(rmax, S[nl * 33 + t] * scale);
        float mnew = fmaxf(mrun, rmax);
        float corr = __expf(mrun - mnew);
        float psum = 0.f;
        bf16x16 pb;
#pragma unroll
        for (int t = 0; t < 32; ++t) {
            float p = (t < kvalid) ? __expf(S[nl * 33 + t] * scale - mnew) : 0.f;
            psum += p;
            if ((t >> 4) == half) pb[t & 15] = (__bf16)p;
        }
        lrun = lrun * corr + psum;
        mrun = mnew;
        o0 *= corr;
        o1 *= corr;

        // V fragments: aligned 32B loads (padding region is zero-filled)
        bf16x16 av0 = *(const bf16x16*)(vb + (size_t)nl * NKP + kb0 + half * 16);
        bf16x16 av1 = *(const bf16x16*)(vb + (size_t)(16 + nl) * NKP + kb0 + half * 16);
        o0 = wmma_bf16(av0, pb, o0);
        o1 = wmma_bf16(av1, pb, o1);
    }

    float inv = 1.f / lrun;
    __bf16* ob = obf + (size_t)b * CC * N + ((size_t)head * N + q0 + nl) * 32;
#pragma unroll
    for (int t2 = 0; t2 < 2; ++t2) {
        bf16x8 pk;
#pragma unroll
        for (int r = 0; r < 8; ++r)
            pk[r] = (__bf16)((t2 ? o1[r] : o0[r]) * inv);
        *(bf16x8*)(ob + t2 * 16 + 8 * half) = pk;
    }
}

__global__ void k_add(const float* __restrict__ a, const float* __restrict__ b,
                      float* __restrict__ o, int n) {
    int t = blockIdx.x * blockDim.x + threadIdx.x;
    if (t < n) o[t] = a[t] + b[t];
}

// ---------------------------------------------------------------------------
extern "C" void kernel_launch(void* const* d_in, const int* in_sizes, int n_in,
                              void* d_out, int out_size, void* d_ws, size_t ws_size,
                              hipStream_t stream) {
    const float* x        = (const float*)d_in[0];
    const float* g1       = (const float*)d_in[1];
    const float* b1       = (const float*)d_in[2];
    const float* g2       = (const float*)d_in[3];
    const float* b2       = (const float*)d_in[4];
    const float* g3       = (const float*)d_in[5];
    const float* b3       = (const float*)d_in[6];
    const float* w_red    = (const float*)d_in[7];    // [3,64,256] == [192,256]
    const float* b_red    = (const float*)d_in[8];
    const float* w_ms3    = (const float*)d_in[9];
    const float* b_ms3    = (const float*)d_in[10];
    const float* w_ms5    = (const float*)d_in[11];
    const float* b_ms5    = (const float*)d_in[12];
    const float* w_ms7    = (const float*)d_in[13];
    const float* b_ms7    = (const float*)d_in[14];
    const float* w_pw     = (const float*)d_in[15];
    const float* b_pw     = (const float*)d_in[16];
    const float* gm       = (const float*)d_in[17];
    const float* bm       = (const float*)d_in[18];
    const float* w_eca    = (const float*)d_in[19];
    const float* wq       = (const float*)d_in[20];
    const float* wk       = (const float*)d_in[21];
    const float* wv       = (const float*)d_in[22];
    const float* wo       = (const float*)d_in[23];
    const float* off_dw_w = (const float*)d_in[24];
    const float* off_dw_b = (const float*)d_in[25];
    const float* off_pw_w = (const float*)d_in[26];
    const float* off_pw_b = (const float*)d_in[27];
    const float* w_dw     = (const float*)d_in[28];
    const float* b_dw     = (const float*)d_in[29];
    const float* w_f1     = (const float*)d_in[30];
    const float* b_f1     = (const float*)d_in[31];
    const float* w_f2     = (const float*)d_in[32];
    const float* b_f2     = (const float*)d_in[33];
    (void)in_sizes; (void)n_in; (void)out_size; (void)ws_size;

    const size_t S  = (size_t)BB * CC * NN;    // 3,211,264
    const size_t SK = (size_t)BB * CC * NKV;   //   802,816
    const size_t SV = (size_t)BB * CC * NKP;   //   819,200

    char* wp = (char*)d_ws;
    auto alloc = [&](size_t bytes) -> void* {
        void* r = (void*)wp;
        wp += (bytes + 255) & ~(size_t)255;
        return r;
    };
    float* F0   = (float*)alloc(S * 4);                 // n1 / o / ln2 / ln3
    float* F1   = (float*)alloc(S * 4);                 // p_all / q (f32)
    float* F2   = (float*)alloc(S * 4);                 // y / x3
    float* F3   = (float*)alloc(S * 4);                 // x2
    float* off1 = (float*)alloc(SK * 4);
    float* off2 = (float*)alloc((size_t)BB * GG * 2 * NKV * 4);
    float* pool = (float*)alloc((size_t)BB * CC * 4);
    float* attb = (float*)alloc((size_t)BB * CC * 4);
    __bf16* n1bf   = (__bf16*)alloc(S * 2);             // B-pack
    __bf16* ybf    = (__bf16*)alloc(S * 2);             // B-pack
    __bf16* sampbf = (__bf16*)alloc(SK * 2);            // B-pack
    __bf16* aobf   = (__bf16*)alloc(S * 2);             // B-pack
    __bf16* ln3bf  = (__bf16*)alloc(S * 2);             // B-pack
    __bf16* h1bf   = (__bf16*)alloc((size_t)BB * 1024 * NN * 2);   // B-pack
    __bf16* qa     = (__bf16*)alloc(S * 2);             // A-pack channel-last
    __bf16* kpbf   = (__bf16*)alloc(SK * 2);            // B-pack
    __bf16* vabf   = (__bf16*)alloc(SV * 2);            // row-major K-perm, padded
    __bf16* wpR  = (__bf16*)alloc((size_t)192 * 256 * 2);
    __bf16* wpP  = (__bf16*)alloc((size_t)64 * 256 * 2);
    __bf16* wpQ  = (__bf16*)alloc((size_t)256 * 256 * 2);
    __bf16* wpK  = (__bf16*)alloc((size_t)256 * 256 * 2);
    __bf16* wpV  = (__bf16*)alloc((size_t)256 * 256 * 2);
    __bf16* wpO  = (__bf16*)alloc((size_t)256 * 256 * 2);
    __bf16* wpF1 = (__bf16*)alloc((size_t)1024 * 256 * 2);
    __bf16* wpF2 = (__bf16*)alloc((size_t)256 * 1024 * 2);

    const int T = 256;
    dim3 blk(128);

    // 0) weight packing (f32 -> bf16 A-pack)
    k_packA<<<(192 * 8 + T - 1) / T, T, 0, stream>>>(w_red, wpR, 192, 256);
    k_packA<<<(64 * 8 + T - 1) / T, T, 0, stream>>>(w_pw, wpP, 64, 256);
    k_packA<<<(256 * 8 + T - 1) / T, T, 0, stream>>>(wq, wpQ, 256, 256);
    k_packA<<<(256 * 8 + T - 1) / T, T, 0, stream>>>(wk, wpK, 256, 256);
    k_packA<<<(256 * 8 + T - 1) / T, T, 0, stream>>>(wv, wpV, 256, 256);
    k_packA<<<(256 * 8 + T - 1) / T, T, 0, stream>>>(wo, wpO, 256, 256);
    k_packA<<<(1024 * 8 + T - 1) / T, T, 0, stream>>>(w_f1, wpF1, 1024, 256);
    k_packA<<<(256 * 32 + T - 1) / T, T, 0, stream>>>(w_f2, wpF2, 256, 1024);
    k_zero<<<((int)SV + T - 1) / T, T, 0, stream>>>(vabf, (int)SV);

    // 1) LN1 -> F0 (f32) + n1bf (B-pack)
    k_ln<<<(BB * NN + T - 1) / T, T, 0, stream>>>(x, g1, b1, F0, n1bf, BB, CC, NN);

    // 2) MSPU pointwise
    k_gemm<<<dim3(49, 3, BB), blk, 0, stream>>>(wpR, n1bf, b_red, nullptr, F1,
                                                nullptr, 192, CC, NN, CC, 0, 0, 0, NN);
    k_gemm<<<dim3(49, 1, BB), blk, 0, stream>>>(wpP, n1bf, b_pw, nullptr, F1,
                                                nullptr, 64, CC, NN, CC, 192, 0, 0, NN);

    // 3) depthwise branches dw(p)+p -> F2; pw branch copied
    const float* wms[3] = {w_ms3, w_ms5, w_ms7};
    const float* bms[3] = {b_ms3, b_ms5, b_ms7};
    for (int i = 0; i < 3; ++i) {
        int k = 3 + 2 * i;
        k_dw<<<((int)(BB * 64 * NN) + T - 1) / T, T, 0, stream>>>(
            F1, wms[i], bms[i], F1, F2, BB, CC, i * 64, 64, HH, WW, HH, WW, k, 1,
            k / 2, 0);
    }
    k_copyc<<<((int)(BB * 64 * NN) + T - 1) / T, T, 0, stream>>>(F1, F2, BB, CC,
                                                                 192, 64, NN);

    // 4) LN(gm,bm) in place
    k_ln<<<(BB * NN + T - 1) / T, T, 0, stream>>>(F2, gm, bm, F2, nullptr, BB, CC, NN);

    // 5) ECA
    k_mean<<<(BB * CC + 127) / 128, 128, 0, stream>>>(F2, pool, BB * CC, NN);
    k_eca<<<(BB * CC + 127) / 128, 128, 0, stream>>>(pool, w_eca, attb);
    k_scale<<<((int)S + T - 1) / T, T, 0, stream>>>(F2, attb, ybf, NN, (int)S);

    // 6) q projection: f32 (for offsets) + A-pack (for attention)
    k_gemm<<<dim3(49, 4, BB), blk, 0, stream>>>(wpQ, ybf, nullptr, nullptr, F1, qa,
                                                CC, CC, NN, CC, 0, 0, 1, NN);

    // 7) offsets + bilinear sampling
    k_dw<<<((int)(BB * GG * CGC * NKV) + T - 1) / T, T, 0, stream>>>(
        F1, off_dw_w, off_dw_b, nullptr, off1, BB * GG, CGC, 0, CGC, HH, WW, HK, WK,
        5, 2, 2, 1);
    k_pwoff<<<((int)(BB * GG * 2 * NKV) + T - 1) / T, T, 0, stream>>>(off1, off_pw_w,
                                                                      off_pw_b, off2);
    k_sample<<<((int)(BB * GG * NKV) + T - 1) / T, T, 0, stream>>>(F2, off2, sampbf);

    // 8) k/v projections -> packed bf16 only
    k_gemm<<<dim3(13, 4, BB), blk, 0, stream>>>(wpK, sampbf, nullptr, nullptr,
                                                nullptr, kpbf, CC, CC, NKV, CC, 0, 0,
                                                0, NKV);
    k_gemm<<<dim3(13, 4, BB), blk, 0, stream>>>(wpV, sampbf, nullptr, nullptr,
                                                nullptr, vabf, CC, CC, NKV, CC, 0, 0,
                                                2, NKP);

    // 9) flash attention -> aobf (B-pack)
    k_attn<<<dim3(49, HEADS, BB), blk, 0, stream>>>(qa, kpbf, vabf, aobf, NN, NKV);

    // 10) output projection + residual chain
    k_gemm<<<dim3(49, 4, BB), blk, 0, stream>>>(wpO, aobf, nullptr, nullptr, F0,
                                                nullptr, CC, CC, NN, CC, 0, 0, 0, NN);
    k_add<<<((int)S + T - 1) / T, T, 0, stream>>>(x, F0, F3, (int)S);   // x2

    k_ln<<<(BB * NN + T - 1) / T, T, 0, stream>>>(F3, g2, b2, F0, nullptr, BB, CC, NN);
    k_dw<<<((int)S + T - 1) / T, T, 0, stream>>>(F0, w_dw, b_dw, F3, F2, BB, CC, 0,
                                                 CC, HH, WW, HH, WW, 3, 1, 1, 0); // x3

    // 11) FFN
    k_ln<<<(BB * NN + T - 1) / T, T, 0, stream>>>(F2, g3, b3, F0, ln3bf, BB, CC, NN);
    k_gemm<<<dim3(49, 16, BB), blk, 0, stream>>>(wpF1, ln3bf, b_f1, nullptr, nullptr,
                                                 h1bf, 1024, CC, NN, 1024, 0, 1, 0, NN);
    k_gemm<<<dim3(49, 4, BB), blk, 0, stream>>>(wpF2, h1bf, b_f2, F2, (float*)d_out,
                                                nullptr, CC, 1024, NN, CC, 0, 0, 0, NN);
}
